// WanSelfAttention_89137751261467
// MI455X (gfx1250) — compile-verified, matrix operand order checked by reading, not run on hardware
//
#include <hip/hip_runtime.h>
#include <stdint.h>

#define DIMN 3072
#define NHEADS 24
#define HEAD_DIM 128
#define CPAIR 64           // rope pairs per head
#define BB 2
#define LLEN 2048
#define MTOK (BB * LLEN)   // 4096
#define RC1 22
#define RC2 21
#define EPSF 1e-6f

typedef __attribute__((ext_vector_type(16))) __bf16 v16bf;
typedef __attribute__((ext_vector_type(8)))  float  v8f;
typedef __attribute__((ext_vector_type(4)))  unsigned int v4u;
typedef __attribute__((ext_vector_type(8)))  int    v8i;
typedef __attribute__((ext_vector_type(4)))  int    v4i;

union ABf { v16bf v; uint4 q[2]; unsigned short u[16]; };

__device__ __forceinline__ unsigned short f2bf(float f) {
  unsigned int u = __float_as_uint(f);
  unsigned int r = u + 0x7FFFu + ((u >> 16) & 1u);
  return (unsigned short)(r >> 16);
}
__device__ __forceinline__ float bf2f(unsigned short h) {
  return __uint_as_float(((unsigned int)h) << 16);
}
__device__ __forceinline__ v8f vzero8() {
  v8f z = {0.f, 0.f, 0.f, 0.f, 0.f, 0.f, 0.f, 0.f};
  return z;
}
// low 32 bits of a generic pointer to __shared__ = group-segment byte offset
__device__ __forceinline__ unsigned lds_off(const void* p) {
  return (unsigned)(uintptr_t)p;
}
// async per-lane global->LDS copy of 16 bytes (ASYNCcnt tracked)
__device__ __forceinline__ void async_g2l_b128(unsigned lds_addr,
                                               const void* gaddr) {
  asm volatile("global_load_async_to_lds_b128 %0, %1, off"
               :
               : "v"(lds_addr), "v"(gaddr)
               : "memory");
}
__device__ __forceinline__ void wait_async0() {
  asm volatile("s_wait_asynccnt 0x0" ::: "memory");
}

// ---------------------------------------------------------------- convert
__global__ void cvt_f32_bf16(const float* __restrict__ in,
                             unsigned short* __restrict__ out, int n) {
  int i = blockIdx.x * blockDim.x + threadIdx.x;
  int stride = gridDim.x * blockDim.x;
  for (; i < n; i += stride) out[i] = f2bf(in[i]);
}

// ---------------------------------------------------------------- GEMM
// C[M,N] = A[M,K](bf16) * Bw[K,N](bf16) + bias[N]; out bf16 or f32.
// Workgroup tile 128(M) x 128(N), 8 waves: wave grid 4(M) x 2(N),
// wave tile 32x64 -> 8 WMMAs per K=32 step. Double-buffered LDS:
// A tile staged via global_load_async_to_lds_b128 overlapping WMMA,
// B tile transposed through LDS. One barrier per K step.
__global__ __launch_bounds__(256)
void gemm_bf16(const unsigned short* __restrict__ A,
               const unsigned short* __restrict__ Bw,
               const float* __restrict__ bias,
               void* __restrict__ Out,
               int M, int N, int K, int out_f32) {
  __shared__ __align__(16) unsigned short sA[2][128][48];  // [m][k]
  __shared__ __align__(16) unsigned short sB[2][128][48];  // [n][k] (transp.)
  (void)M;

  const int tid   = threadIdx.x;
  const int lane  = tid & 31;
  const int wid   = tid >> 5;
  const int waveM = wid & 3;          // *32 rows
  const int waveN = wid >> 2;         // *64 cols
  const int l16   = lane & 15;
  const int hsel  = lane >> 4;        // 0 / 1
  const int kbase = hsel << 3;        // 0 / 8  (16-bit A-layout K split)
  const int m0g   = blockIdx.x * 128;
  const int n0g   = blockIdx.y * 128;

  const int arow = tid >> 1;             // 0..127
  const int acol = (tid & 1) << 3;       // 0 / 8 (then +16 for second issue)
  const int bks  = tid & 31;             // k within slab
  const int bng  = (tid >> 5) << 4;      // n group 0..112

  auto stage = [&](int buf, int kk) {
    // A tile (128x32): 2 x 16B async global->LDS per thread
    const unsigned short* ap = &A[(size_t)(m0g + arow) * K + kk + acol];
    async_g2l_b128(lds_off(&sA[buf][arow][acol]), ap);
    async_g2l_b128(lds_off(&sA[buf][arow][acol + 16]), ap + 16);
    // B tile (32x128) transposed -> sB[buf][n][k]
    const unsigned short* bp = &Bw[(size_t)(kk + bks) * N + n0g + bng];
    uint4 b0 = *(const uint4*)bp;
    uint4 b1 = *(const uint4*)(bp + 8);
    const unsigned short* bu0 = (const unsigned short*)&b0;
    const unsigned short* bu1 = (const unsigned short*)&b1;
#pragma unroll
    for (int e = 0; e < 8; ++e) {
      sB[buf][bng + e][bks]     = bu0[e];
      sB[buf][bng + 8 + e][bks] = bu1[e];
    }
    if (kk + 32 < K) {  // prefetch slab after next (global_prefetch_b8)
      __builtin_prefetch(&A[(size_t)(m0g + arow) * K + kk + 32 + acol], 0, 1);
      __builtin_prefetch(&Bw[(size_t)(kk + 32 + bks) * N + n0g + bng], 0, 1);
    }
  };

  v8f acc[2][4];
#pragma unroll
  for (int mf = 0; mf < 2; ++mf)
#pragma unroll
    for (int j = 0; j < 4; ++j) acc[mf][j] = vzero8();

  // prologue: stage first tile
  stage(0, 0);
  wait_async0();
  __syncthreads();

  int buf = 0;
  for (int kk = 0; kk < K; kk += 32) {
    // stage next tile into the other buffer (overlaps the WMMA burst)
    if (kk + 32 < K) stage(buf ^ 1, kk + 32);

    // preload all fragments, then burst 8 WMMAs
    ABf aF[2];
#pragma unroll
    for (int mf = 0; mf < 2; ++mf) {
      const int mr = waveM * 32 + mf * 16 + l16;
      aF[mf].q[0] = *(const uint4*)&sA[buf][mr][kbase];
      aF[mf].q[1] = *(const uint4*)&sA[buf][mr][kbase + 16];
    }
    ABf bF[4];
#pragma unroll
    for (int j = 0; j < 4; ++j) {
      const int nl = waveN * 64 + j * 16 + l16;
      bF[j].q[0] = *(const uint4*)&sB[buf][nl][kbase];
      bF[j].q[1] = *(const uint4*)&sB[buf][nl][kbase + 16];
    }
#pragma unroll
    for (int j = 0; j < 4; ++j) {
#pragma unroll
      for (int mf = 0; mf < 2; ++mf) {
        acc[mf][j] = __builtin_amdgcn_wmma_f32_16x16x32_bf16(
            false, aF[mf].v, false, bF[j].v, (short)0, acc[mf][j], false,
            false);
      }
    }
    wait_async0();    // next tile landed in LDS
    __syncthreads();  // everyone done reading buf / staging buf^1
    buf ^= 1;
  }

  // epilogue: C layout = VGPR r: lanes0-15 -> M=r, lanes16-31 -> M=r+8
#pragma unroll
  for (int mf = 0; mf < 2; ++mf) {
#pragma unroll
    for (int j = 0; j < 4; ++j) {
      const int n = n0g + waveN * 64 + j * 16 + l16;
      const float bb = bias[n];
#pragma unroll
      for (int r = 0; r < 8; ++r) {
        const int m = m0g + waveM * 32 + mf * 16 + r + (hsel << 3);
        const float val = acc[mf][j][r] + bb;
        if (out_f32) ((float*)Out)[(size_t)m * N + n] = val;
        else ((unsigned short*)Out)[(size_t)m * N + n] = f2bf(val);
      }
    }
  }
}

// ---------------------------------------------------------------- rmsnorm + rope
// one block per token row (3072 bf16, in-place)
__global__ __launch_bounds__(256)
void rmsrope(unsigned short* __restrict__ qk, const float* __restrict__ g,
             const float* __restrict__ freqs, const int* __restrict__ seq_lens,
             const int* __restrict__ grid_sizes) {
  __shared__ float red[256];
  const int tid = threadIdx.x;
  const int row = blockIdx.x;
  const int b = row / LLEN;
  const int s = row - b * LLEN;
  unsigned short* rp = qk + (size_t)row * DIMN;

  float fr[6], fi[6];
  float ss = 0.f;
#pragma unroll
  for (int i = 0; i < 6; ++i) {
    const int p = tid + i * 256;
    const unsigned int packed = *(const unsigned int*)&rp[2 * p];
    const float xr = bf2f((unsigned short)(packed & 0xFFFFu));
    const float xi = bf2f((unsigned short)(packed >> 16));
    fr[i] = xr; fi[i] = xi;
    ss += xr * xr + xi * xi;
  }
  red[tid] = ss;
  __syncthreads();
  for (int off = 128; off > 0; off >>= 1) {
    if (tid < off) red[tid] += red[tid + off];
    __syncthreads();
  }
  const float rs = rsqrtf(red[0] * (1.0f / DIMN) + EPSF);

  const int seqlen = seq_lens[b];
  const int gh = grid_sizes[b * 3 + 1];
  const int gw = grid_sizes[b * 3 + 2];
  const int hw = gh * gw;
  int zf = 0, zh = 0, zw = 0;
  const bool valid = (s < seqlen);
  if (valid) {
    zf = s / hw;
    const int rem = s - zf * hw;
    zh = rem / gw;
    zw = rem - zh * gw;
  }

#pragma unroll
  for (int i = 0; i < 6; ++i) {
    const int p = tid + i * 256;
    const int c = p & (CPAIR - 1);
    float cs = 1.f, sn = 0.f;
    if (valid) {
      const int pos = (c < RC1) ? zf : ((c < RC1 + RC2) ? zh : zw);
      const float* fp = &freqs[((size_t)pos * CPAIR + c) * 2];
      cs = fp[0]; sn = fp[1];
    }
    const float xr = fr[i] * rs * g[2 * p];
    const float xi = fi[i] * rs * g[2 * p + 1];
    const float outr = xr * cs - xi * sn;
    const float outi = xr * sn + xi * cs;
    const unsigned int packed =
        (unsigned int)f2bf(outr) | ((unsigned int)f2bf(outi) << 16);
    *(unsigned int*)&rp[2 * p] = packed;
  }
}

// ---------------------------------------------------------------- flash attention
// grid: (L/64) * NHEADS * B blocks, 128 threads (4 waves, 16 query rows each).
// Double-buffered: TDM loads the next K tile (D# pads 256B rows by 16B to
// land on the sK[.][64][136] layout) while WMMAs run on the current tile.
__global__ __launch_bounds__(128)
void attention(const unsigned short* __restrict__ q,
               const unsigned short* __restrict__ k,
               const unsigned short* __restrict__ v,
               unsigned short* __restrict__ o,
               const int* __restrict__ seq_lens) {
  __shared__ __align__(16) unsigned short sK[2][64][136];   // [key][d]
  __shared__ __align__(16) unsigned short sVt[2][128][72];  // [d][key]
  __shared__ __align__(16) unsigned short sP[4][16][72];    // per-wave P

  const int tid  = threadIdx.x;
  const int lane = tid & 31;
  const int wid  = tid >> 5;
  const int l16  = lane & 15;
  const int hsel = lane >> 4;
  const int kbase = hsel << 3;
  int bid = blockIdx.x;
  const int mt = bid & 31;           bid >>= 5;
  const int h  = bid % NHEADS;
  const int b  = bid / NHEADS;
  const int seqlen = seq_lens[b];
  const int s0 = mt * 64 + wid * 16;
  const float scale = 0.08838834764831845f;  // 1/sqrt(128)

  const int nkt  = (seqlen + 63) >> 6;
  const int srow = tid >> 1;          // 0..63
  const int sdg  = (tid & 1) << 6;    // 0 / 64

  // stage key/value tile t0 into buffer bf (TDM for K, manual transpose for V)
  auto stage_kv = [&](int bf, int t0) {
#if __has_builtin(__builtin_amdgcn_tensor_load_to_lds)
    if (wid == 0) {
      const unsigned short* kbaseptr =
          k + ((size_t)(b * LLEN + t0)) * DIMN + h * HEAD_DIM;
      const unsigned long long ga = (unsigned long long)(uintptr_t)kbaseptr;
      v4u g0;
      g0.x = 1u;                                   // count=1, user mode
      g0.y = lds_off(&sK[bf][0][0]);               // lds_addr
      g0.z = (unsigned)(ga & 0xFFFFFFFFu);         // global_addr[31:0]
      g0.w = (unsigned)((ga >> 32) & 0x01FFFFFFu)  // global_addr[56:32]
             | (2u << 30);                         // type = 2 (image)
      v8i g1;
      g1[0] = (int)((1u << 16)      // data_size = 2 bytes
                    | (1u << 20)    // pad_enable
                    | (5u << 22)    // pad_interval: every 64 DWORDs (256B)
                    | (3u << 25));  // pad_amount: 4 DWORDs (16B)
      g1[1] = (int)(128u << 16);    // tensor_dim0 = 128
      g1[2] = (int)(64u << 16);     // tensor_dim1 = 64
      g1[3] = (int)(128u << 16);    // tile_dim0 = 128
      g1[4] = 64;                   // tile_dim1 = 64 (tile_dim2 = 0)
      g1[5] = DIMN;                 // tensor_dim0_stride = 3072 elements
      g1[6] = 0;
      g1[7] = 0;
      v4i gz4 = {0, 0, 0, 0};
      v8i gz8 = {0, 0, 0, 0, 0, 0, 0, 0};
      __builtin_amdgcn_tensor_load_to_lds(g0, g1, gz4, gz4, gz8, 0);
    }
#else
    {
      const unsigned short* kp =
          k + ((size_t)(b * LLEN + t0 + srow)) * DIMN + h * HEAD_DIM + sdg;
#pragma unroll
      for (int i = 0; i < 8; ++i)
        *(uint4*)&sK[bf][srow][sdg + i * 8] = *(const uint4*)(kp + i * 8);
    }
#endif
    // V tile transposed into LDS
    const unsigned short* vp =
        v + ((size_t)(b * LLEN + t0 + srow)) * DIMN + h * HEAD_DIM + sdg;
#pragma unroll
    for (int i = 0; i < 8; ++i) {
      uint4 vv = *(const uint4*)(vp + i * 8);
      const unsigned short* vu = (const unsigned short*)&vv;
#pragma unroll
      for (int e = 0; e < 8; ++e) sVt[bf][sdg + i * 8 + e][srow] = vu[e];
    }
  };
  auto wait_tensor = [&]() {
#if __has_builtin(__builtin_amdgcn_tensor_load_to_lds)
    if (wid == 0) __builtin_amdgcn_s_wait_tensorcnt((short)0);
#endif
  };

  // Q A-fragments (rows s0..s0+15, 4 K=32 chunks over head_dim)
  ABf aq[4];
  {
    const unsigned short* qp =
        q + ((size_t)(b * LLEN + s0 + l16)) * DIMN + h * HEAD_DIM;
#pragma unroll
    for (int dc = 0; dc < 4; ++dc) {
      aq[dc].q[0] = *(const uint4*)&qp[dc * 32 + kbase];
      aq[dc].q[1] = *(const uint4*)&qp[dc * 32 + kbase + 16];
    }
  }

  v8f oacc[8];
#pragma unroll
  for (int n = 0; n < 8; ++n) oacc[n] = vzero8();
  float mi[8], li[8];
#pragma unroll
  for (int r = 0; r < 8; ++r) { mi[r] = -3.0e38f; li[r] = 0.f; }

  // prologue: stage first key tile
  stage_kv(0, 0);
  wait_tensor();
  __syncthreads();

  int bf = 0;
  for (int kt = 0; kt < nkt; ++kt) {
    const int t0 = kt * 64;
    // stage next key tile into the other buffer (overlaps compute)
    if (kt + 1 < nkt) stage_kv(bf ^ 1, t0 + 64);

    // S = Q * K^T  (16 rows x 64 keys): 16 WMMAs
    v8f sacc[4];
#pragma unroll
    for (int j = 0; j < 4; ++j) sacc[j] = vzero8();
#pragma unroll
    for (int j = 0; j < 4; ++j) {
      const int kcol = j * 16 + l16;
#pragma unroll
      for (int dc = 0; dc < 4; ++dc) {
        ABf bK;
        bK.q[0] = *(const uint4*)&sK[bf][kcol][dc * 32 + kbase];
        bK.q[1] = *(const uint4*)&sK[bf][kcol][dc * 32 + kbase + 16];
        sacc[j] = __builtin_amdgcn_wmma_f32_16x16x32_bf16(
            false, aq[dc].v, false, bK.v, (short)0, sacc[j], false, false);
      }
    }

    // online softmax update (row = C-layout VGPR index, halves of wave)
#pragma unroll
    for (int r = 0; r < 8; ++r) {
      float mx = -3.0e38f;
#pragma unroll
      for (int j = 0; j < 4; ++j) {
        const int col = t0 + j * 16 + l16;
        const float val = (col < seqlen) ? sacc[j][r] * scale : -1.0e9f;
        sacc[j][r] = val;
        mx = fmaxf(mx, val);
      }
#pragma unroll
      for (int xm = 1; xm < 16; xm <<= 1) mx = fmaxf(mx, __shfl_xor(mx, xm, 32));
      const float mnew = fmaxf(mi[r], mx);
      const float fac = __expf(mi[r] - mnew);
      mi[r] = mnew;
      li[r] *= fac;
#pragma unroll
      for (int n = 0; n < 8; ++n) oacc[n][r] *= fac;
      float rsum = 0.f;
#pragma unroll
      for (int j = 0; j < 4; ++j) {
        const float pv = __expf(sacc[j][r] - mnew);
        sacc[j][r] = pv;
        rsum += pv;
      }
#pragma unroll
      for (int xm = 1; xm < 16; xm <<= 1) rsum += __shfl_xor(rsum, xm, 32);
      li[r] += rsum;
    }

    // restage P (C layout -> bf16 LDS -> A layout), per-wave region
#pragma unroll
    for (int j = 0; j < 4; ++j)
#pragma unroll
      for (int r = 0; r < 8; ++r)
        sP[wid][r + (hsel << 3)][j * 16 + l16] = f2bf(sacc[j][r]);

    // O += P * V : 16 WMMAs
#pragma unroll
    for (int kc = 0; kc < 2; ++kc) {
      ABf aP;
      aP.q[0] = *(const uint4*)&sP[wid][l16][kc * 32 + kbase];
      aP.q[1] = *(const uint4*)&sP[wid][l16][kc * 32 + kbase + 16];
#pragma unroll
      for (int n = 0; n < 8; ++n) {
        ABf bV;
        const int d = n * 16 + l16;
        bV.q[0] = *(const uint4*)&sVt[bf][d][kc * 32 + kbase];
        bV.q[1] = *(const uint4*)&sVt[bf][d][kc * 32 + kbase + 16];
        oacc[n] = __builtin_amdgcn_wmma_f32_16x16x32_bf16(
            false, aP.v, false, bV.v, (short)0, oacc[n], false, false);
      }
    }
    wait_tensor();    // next K tile landed (wave 0)
    __syncthreads();  // everyone done with buf; staging of buf^1 landed
    bf ^= 1;
  }

  // normalize + store bf16
#pragma unroll
  for (int r = 0; r < 8; ++r) {
    const float inv = 1.0f / li[r];
    const int m = r + (hsel << 3);
    const size_t base =
        ((size_t)(b * LLEN + s0 + m)) * DIMN + h * HEAD_DIM;
#pragma unroll
    for (int n = 0; n < 8; ++n)
      o[base + n * 16 + l16] = f2bf(oacc[n][r] * inv);
  }
}

// ---------------------------------------------------------------- launch
extern "C" void kernel_launch(void* const* d_in, const int* in_sizes, int n_in,
                              void* d_out, int out_size, void* d_ws, size_t ws_size,
                              hipStream_t stream) {
  const float* x          = (const float*)d_in[0];
  const int*   seq_lens   = (const int*)d_in[1];
  const int*   grid_sizes = (const int*)d_in[2];
  const float* freqs      = (const float*)d_in[3];
  const float* wq = (const float*)d_in[4];
  const float* bq = (const float*)d_in[5];
  const float* wk = (const float*)d_in[6];
  const float* bk = (const float*)d_in[7];
  const float* wv = (const float*)d_in[8];
  const float* bv = (const float*)d_in[9];
  const float* wo = (const float*)d_in[10];
  const float* bo = (const float*)d_in[11];
  const float* gq = (const float*)d_in[12];
  const float* gk = (const float*)d_in[13];
  float* out = (float*)d_out;

  char* ws = (char*)d_ws;
  const size_t XB = (size_t)MTOK * DIMN * sizeof(unsigned short); // 25.2 MB
  const size_t WB = (size_t)DIMN * DIMN * sizeof(unsigned short); // 18.9 MB
  size_t off = 0;
  unsigned short* xb  = (unsigned short*)(ws + off); off += XB;
  unsigned short* wqb = (unsigned short*)(ws + off); off += WB;
  unsigned short* wkb = (unsigned short*)(ws + off); off += WB;
  unsigned short* wvb = (unsigned short*)(ws + off); off += WB;
  unsigned short* wob = (unsigned short*)(ws + off); off += WB;
  unsigned short* qb  = (unsigned short*)(ws + off); off += XB;
  unsigned short* kb  = (unsigned short*)(ws + off); off += XB;
  unsigned short* vb  = (unsigned short*)(ws + off); off += XB;
  unsigned short* ab  = xb;  // x no longer needed after QKV GEMMs

  cvt_f32_bf16<<<2048, 256, 0, stream>>>(x, xb, MTOK * DIMN);
  cvt_f32_bf16<<<2048, 256, 0, stream>>>(wq, wqb, DIMN * DIMN);
  cvt_f32_bf16<<<2048, 256, 0, stream>>>(wk, wkb, DIMN * DIMN);
  cvt_f32_bf16<<<2048, 256, 0, stream>>>(wv, wvb, DIMN * DIMN);
  cvt_f32_bf16<<<2048, 256, 0, stream>>>(wo, wob, DIMN * DIMN);

  dim3 gg(MTOK / 128, DIMN / 128);
  gemm_bf16<<<gg, 256, 0, stream>>>(xb, wqb, bq, qb, MTOK, DIMN, DIMN, 0);
  gemm_bf16<<<gg, 256, 0, stream>>>(xb, wkb, bk, kb, MTOK, DIMN, DIMN, 0);
  gemm_bf16<<<gg, 256, 0, stream>>>(xb, wvb, bv, vb, MTOK, DIMN, DIMN, 0);

  rmsrope<<<MTOK, 256, 0, stream>>>(qb, gq, freqs, seq_lens, grid_sizes);
  rmsrope<<<MTOK, 256, 0, stream>>>(kb, gk, freqs, seq_lens, grid_sizes);

  attention<<<BB * NHEADS * (LLEN / 64), 128, 0, stream>>>(qb, kb, vb, ab,
                                                           seq_lens);

  gemm_bf16<<<gg, 256, 0, stream>>>(ab, wob, bo, out, MTOK, DIMN, DIMN, 1);
}
